// MultiAreaEpi_46265387712818
// MI455X (gfx1250) — matrix operand order repaired and checked by reading, not compiled, not run on hardware
//
#include <hip/hip_runtime.h>
#include <hip/hip_bf16.h>

// ---------------------------------------------------------------------------
// Types for CDNA5 WMMA (wave32): 16x16x32 bf16 -> f32 accumulate
// ---------------------------------------------------------------------------
typedef __attribute__((ext_vector_type(16))) __bf16 v16bf;
typedef __attribute__((ext_vector_type(2)))  __bf16 v2bf;
typedef __attribute__((ext_vector_type(8)))  float  v8f;

union AView {
    v16bf    v;
    unsigned w[8];
};

static __device__ __forceinline__ __bf16 f2bf(float f) {
    return (__bf16)f;   // native v_cvt on gfx1250
}

// Pack two floats into one dword of 2 x bf16 (lo in [15:0], hi in [31:16])
static __device__ __forceinline__ unsigned pkbf(float lo, float hi) {
#if __has_builtin(__builtin_amdgcn_cvt_pk_bf16_f32)
    v2bf p = __builtin_amdgcn_cvt_pk_bf16_f32(lo, hi);
    return __builtin_bit_cast(unsigned, p);
#else
    unsigned short a = __builtin_bit_cast(unsigned short, (__bf16)lo);
    unsigned short b = __builtin_bit_cast(unsigned short, (__bf16)hi);
    return (unsigned)a | ((unsigned)b << 16);
#endif
}

// Pure-VALU cross-lane move via DPP (no ds_bpermute)
template <int CTRL>
static __device__ __forceinline__ float dppf(float v) {
    return __builtin_bit_cast(float, __builtin_amdgcn_update_dpp(
        0, __builtin_bit_cast(int, v), CTRL, 0xF, 0xF, true));
}
#define DPP_QUAD_XOR1 0xB1   // quad_perm [1,0,3,2]
#define DPP_ROW_SHR1  0x111
#define DPP_ROW_SHR2  0x112
#define DPP_ROW_SHR4  0x114
#define DPP_ROW_SHR8  0x118

#define NB   2048
#define NC   200
#define NF   32
#define NH   256
#define NROWS (NB * NC)          // 409600
#define NTILES (NROWS / 16)      // 25600
#define GRID1 3200               // must be <= NTILES

// ---------------------------------------------------------------------------
// Kernel 1: fused 3-layer MLP, lamb[b*C+c] = W3.relu(W2.relu(W1.x+b1)+b2)+b3
// Each wave owns a 32-col slice of H; W1/W2 slices pre-swizzled into bf16
// B-operand register layout (W2 stays resident in VGPRs across all tiles).
// Layer-1 A double-buffered in registers (next tile's load issued in the
// WMMA shadow); h1 staged in LDS in the WMMA 16-bit A layout.
// ---------------------------------------------------------------------------
__global__ void __launch_bounds__(256, 1)
mlp_lambda_kernel(const float* __restrict__ x,
                  const float* __restrict__ W1, const float* __restrict__ b1,
                  const float* __restrict__ W2, const float* __restrict__ b2,
                  const float* __restrict__ W3, const float* __restrict__ b3,
                  float* __restrict__ lamb)
{
    const int lane = threadIdx.x & 31;
    const int wave = threadIdx.x >> 5;     // 0..7
    const int lcol = lane & 15;            // N index within 16-wide tile
    const int hi   = lane >> 4;            // 0 / 1 (lane half)
    const int n0   = wave * 32;            // this wave's H-column base

    // --- per-lane epilogue scalars -------------------------------------
    const float b1v0 = b1[n0 + lcol];
    const float b1v1 = b1[n0 + 16 + lcol];
    const float b2v0 = b2[n0 + lcol];
    const float b2v1 = b2[n0 + 16 + lcol];
    const float w3v0 = W3[n0 + lcol];
    const float w3v1 = W3[n0 + 16 + lcol];
    const float b3v  = b3[0];

    // --- preload W1 slice as 2 bf16 B operands -------------------------
    // B layout (16-bit, 32x16): lanes 0-15 = col n, K={0..7,16..23} ; lanes 16-31
    // same cols, K={8..15,24..31} ; 2 bf16 per VGPR in K order.
    v16bf B1[2];
#pragma unroll
    for (int t = 0; t < 2; ++t) {
        const int col   = n0 + t * 16 + lcol;
        const int kbase = hi * 8;
#pragma unroll
        for (int k = 0; k < 16; ++k)
            B1[t][k] = f2bf(W1[(size_t)(kbase + (k & 7) + ((k & 8) << 1)) * NH + col]);
    }

    // --- preload W2 slice as 16 bf16 B operands (resident in VGPRs) ----
    v16bf B2[8][2];
#pragma unroll
    for (int kc = 0; kc < 8; ++kc) {
#pragma unroll
        for (int t = 0; t < 2; ++t) {
            const int col   = n0 + t * 16 + lcol;
            const int kbase = kc * 32 + hi * 8;
#pragma unroll
            for (int k = 0; k < 16; ++k)
                B2[kc][t][k] = f2bf(W2[(size_t)(kbase + (k & 7) + ((k & 8) << 1)) * NH + col]);
        }
    }

    // --- LDS staging buffers -------------------------------------------
    __shared__ alignas(16) __bf16 h1A[8][512];     // h1 tile, 8 K-chunks (8 KB)
    __shared__ float lambP[8][16];                 // per-wave layer-3 partials

    // constants for the layer-1 pair-pack epilogue
    const bool evenLane = (lcol & 1) == 0;
    // even column this lane stores: even lanes -> tile0 pair (lcol, lcol+1);
    // odd lanes -> tile1 pair (16+lcol-1, 16+lcol)
    const int Keven = evenLane ? lcol : (16 + lcol - 1);
    const int posE  = (Keven & 7) + ((Keven & 16) ? 8 : 0);
    const int lrOff = (Keven & 8) ? 16 : 0;

    // A-operand loader: lane l -> row l&15 ; lanes 0-15 hold K={0..7,16..23},
    // lanes 16-31 hold K={8..15,24..31}  (16-bit 16x32 A layout)
    auto loadX = [&](int rowBase) -> AView {
        const float* xr = &x[(size_t)(rowBase + lcol) * NF + hi * 8];
        const float4 xa = *(const float4*)(xr + 0);
        const float4 xb = *(const float4*)(xr + 4);
        const float4 xc = *(const float4*)(xr + 16);
        const float4 xd = *(const float4*)(xr + 20);
        AView a;
        a.w[0] = pkbf(xa.x, xa.y);  a.w[1] = pkbf(xa.z, xa.w);
        a.w[2] = pkbf(xb.x, xb.y);  a.w[3] = pkbf(xb.z, xb.w);
        a.w[4] = pkbf(xc.x, xc.y);  a.w[5] = pkbf(xc.z, xc.w);
        a.w[6] = pkbf(xd.x, xd.y);  a.w[7] = pkbf(xd.z, xd.w);
        return a;
    };

    AView ax = loadX(blockIdx.x * 16);   // prologue load (GRID1 <= NTILES)

    for (int tile = blockIdx.x; tile < NTILES; tile += gridDim.x) {
        const int rowBase = tile * 16;

        // ---- layer 1: h1 = relu(x @ W1 + b1)  (2 WMMAs / wave) --------
        v8f acc0 = {};
        v8f acc1 = {};
        acc0 = __builtin_amdgcn_wmma_f32_16x16x32_bf16(false, ax.v, false, B1[0],
                                                       (short)0, acc0, false, false);
        acc1 = __builtin_amdgcn_wmma_f32_16x16x32_bf16(false, ax.v, false, B1[1],
                                                       (short)0, acc1, false, false);

        // ---- independent work in the WMMA shadow: next tile's A ------
        const int ntile = tile + gridDim.x;
        AView axn = ax;
        if (ntile < NTILES) {
            axn = loadX(ntile * 16);
            // prefetch two tiles ahead (2 KB = 32 cachelines)
            const int ptile = ntile + gridDim.x;
            if (ptile < NTILES && lane == 0)
                __builtin_prefetch(((const char*)&x[(size_t)ptile * 16 * NF]) +
                                   (size_t)wave * 256, 0, 0);
        }

        // bias+relu, pair-pack via DPP xor-1, store dwords into h1A[wave]
        // (this wave's 32 cols are exactly K-chunk `wave` of layer 2).
#pragma unroll
        for (int r = 0; r < 8; ++r) {
            const float v0 = fmaxf(acc0[r] + b1v0, 0.0f);
            const float v1 = fmaxf(acc1[r] + b1v1, 0.0f);
            const float p0 = dppf<DPP_QUAD_XOR1>(v0);   // partner column value
            const float p1 = dppf<DPP_QUAD_XOR1>(v1);
            const float lo = evenLane ? v0 : p1;
            const float hh = evenLane ? p0 : v1;
            const int rowT = r + hi * 8;
            const int lr   = rowT + lrOff;
            *(unsigned*)&h1A[wave][lr * 16 + posE] = pkbf(lo, hh);
        }
        __syncthreads();

        // ---- layer 2: h2 = relu(h1 @ W2 + b2)  (16 WMMAs / wave) ------
        v8f c0 = {};
        v8f c1 = {};
#pragma unroll
        for (int kc = 0; kc < 8; ++kc) {
            AView a2;
            *(uint4*)&a2.w[0] = *(const uint4*)(&h1A[kc][(size_t)lane * 16]);
            *(uint4*)&a2.w[4] = *(const uint4*)(&h1A[kc][(size_t)lane * 16 + 8]);
            c0 = __builtin_amdgcn_wmma_f32_16x16x32_bf16(false, a2.v, false, B2[kc][0],
                                                         (short)0, c0, false, false);
            c1 = __builtin_amdgcn_wmma_f32_16x16x32_bf16(false, a2.v, false, B2[kc][1],
                                                         (short)0, c1, false, false);
        }

        // ---- layer 3: lamb = h2 @ W3 + b3 -----------------------------
        // per-lane dot, then 16-lane row sum via DPP row_shr (0-fill);
        // full sum lands in lane 0 (rows 0-7) and lane 16 (rows 8-15).
#pragma unroll
        for (int r = 0; r < 8; ++r) {
            const float h20 = fmaxf(c0[r] + b2v0, 0.0f);
            const float h21 = fmaxf(c1[r] + b2v1, 0.0f);
            float t = h20 * w3v0 + h21 * w3v1;
            t += dppf<DPP_ROW_SHR1>(t);
            t += dppf<DPP_ROW_SHR2>(t);
            t += dppf<DPP_ROW_SHR4>(t);
            t += dppf<DPP_ROW_SHR8>(t);
            if (lcol == 0)
                lambP[wave][r + hi * 8] = t;   // lane0 -> row r, lane16 -> row r+8
        }
        __syncthreads();

        if (threadIdx.x < 16) {
            float s = b3v;
#pragma unroll
            for (int w = 0; w < 8; ++w) s += lambP[w][threadIdx.x];
            lamb[(size_t)rowBase + threadIdx.x] = s;
        }
        ax = axn;
        __syncthreads();   // protect h1A/lambP for next iteration
    }
}

// ---------------------------------------------------------------------------
// Kernel 2: transition aggregation + SIR update + interleave.
// One block per batch row.  T (160 KB) lives in the 192 MB L2.
// ---------------------------------------------------------------------------
__global__ void __launch_bounds__(256)
combine_kernel(const float* __restrict__ x,
               const float* __restrict__ T,
               const float* __restrict__ lamb,
               float* __restrict__ out)
{
    __shared__ float Ssh[NC];
    __shared__ float Ish[NC];

    const int b   = blockIdx.x;
    const int tid = threadIdx.x;

    for (int i = tid; i < NC; i += 256) {
        Ssh[i] = x[(size_t)b * (NC * NF) + (size_t)i * NF + 0];
        Ish[i] = x[(size_t)b * (NC * NF) + (size_t)i * NF + 1];
    }
    __syncthreads();

    if (tid < NC) {
        const int c = tid;
        float sin = 0.0f, iin = 0.0f;
        for (int j = 0; j < NC; ++j) {
            const float t = T[(size_t)j * NC + c];   // coalesced across c
            sin += Ssh[j] * t;
            iin += Ish[j] * t;
        }
        const float of  = 1.0f - T[(size_t)c * NC + c];
        const float S   = Ssh[c];
        const float I   = Ish[c];
        const float lam = lamb[(size_t)b * NC + c];
        const float dS  = -lam * S + sin - of * S;
        const float dI  =  lam * S + iin - of * I;
        out[(size_t)b * (2 * NC) + 2 * c + 0] = S + dS;
        out[(size_t)b * (2 * NC) + 2 * c + 1] = I + dI;
    }
}

// ---------------------------------------------------------------------------
extern "C" void kernel_launch(void* const* d_in, const int* in_sizes, int n_in,
                              void* d_out, int out_size, void* d_ws, size_t ws_size,
                              hipStream_t stream) {
    (void)in_sizes; (void)n_in; (void)out_size; (void)ws_size;
    const float* x  = (const float*)d_in[0];
    const float* T  = (const float*)d_in[1];
    const float* W1 = (const float*)d_in[2];
    const float* b1 = (const float*)d_in[3];
    const float* W2 = (const float*)d_in[4];
    const float* b2 = (const float*)d_in[5];
    const float* W3 = (const float*)d_in[6];
    const float* b3 = (const float*)d_in[7];
    float* out     = (float*)d_out;
    float* lamb_ws = (float*)d_ws;        // NROWS floats = 1.64 MB scratch

    mlp_lambda_kernel<<<GRID1, 256, 0, stream>>>(x, W1, b1, W2, b2, W3, b3, lamb_ws);
    combine_kernel<<<NB, 256, 0, stream>>>(x, T, lamb_ws, out);
}
